// AttentionOnDetail_85452669321584
// MI455X (gfx1250) — matrix operand order repaired and checked by reading, compiled, not run on hardware
//
#include <hip/hip_runtime.h>
#include <hip/hip_bf16.h>

typedef __attribute__((ext_vector_type(16))) _Float16 v16h;
typedef __attribute__((ext_vector_type(8)))  _Float16 v8h;
typedef __attribute__((ext_vector_type(8)))  float    v8f;

#define PI_F 3.14159265358979f
#define RMS_EPS_F 1.1920928955078125e-07f
#define SCALE_F 0.12f

// ---------------------------------------------------------------------------
// WMMA fragment loaders for V_WMMA_F32_16X16X32_F16 (ISA 7.12.2 layouts).
// A (16x32 f16, row-major [M][K]): lane l -> row l&15;
//   halves a[0..7]  = A[r][kb+0 .. kb+7]   (kb = (l>=16)?8:0)
//   halves a[8..15] = A[r][16+kb .. 16+kb+7]
// B (32x16 f16) built from W stored row-major [N][K] (i.e. B = W^T):
//   lane l -> col l&15; b[0..7] = W[c][kb .. kb+7], b[8..15] = W[c][16+kb .. 16+kb+7] (kb = (l>=16)?8:0)
// C/D (16x16 f32): lane l -> col l&15; acc[v] -> row v + ((l>=16)?8:0)
// All spans are 16-byte aligned -> global_load_b128 pairs.
// ---------------------------------------------------------------------------
__device__ __forceinline__ v16h load_a_frag(const _Float16* __restrict__ A,
                                            int lda, int m0, int k0, int lane) {
  const int r  = lane & 15;
  const int kb = (lane & 16) ? 8 : 0;
  const _Float16* p = A + (size_t)(m0 + r) * lda + (k0 + kb);
  v8h lo = *(const v8h*)(p);
  v8h hi = *(const v8h*)(p + 16);
  v16h a;
#pragma unroll
  for (int i = 0; i < 8; ++i) { a[i] = lo[i]; a[8 + i] = hi[i]; }
  return a;
}

__device__ __forceinline__ v16h load_b_frag(const _Float16* __restrict__ W,
                                            int ldw, int n0, int k0, int lane) {
  const int c  = lane & 15;
  const int kb = (lane & 16) ? 8 : 0;
  const _Float16* p = W + (size_t)(n0 + c) * ldw + (k0 + kb);
  v8h lo = *(const v8h*)(p);
  v8h hi = *(const v8h*)(p + 16);
  v16h b;
#pragma unroll
  for (int i = 0; i < 8; ++i) { b[i] = lo[i]; b[8 + i] = hi[i]; }
  return b;
}

#define WMMA_F16(A_, B_, C_) \
  __builtin_amdgcn_wmma_f32_16x16x32_f16(false, (A_), false, (B_), (short)0, (C_), false, false)

// 2x2 register-blocked k-step: 8 b128 loads feed 4 WMMAs (2x fragment reuse)
__device__ __forceinline__ void gemm_step_2x2(
    const _Float16* __restrict__ A, int lda, int m0,
    const _Float16* __restrict__ W, int ldw, int n0,
    int k0, int lane, v8f& c00, v8f& c01, v8f& c10, v8f& c11) {
  v16h a0 = load_a_frag(A, lda, m0, k0, lane);
  v16h a1 = load_a_frag(A, lda, m0 + 16, k0, lane);
  v16h b0 = load_b_frag(W, ldw, n0, k0, lane);
  v16h b1 = load_b_frag(W, ldw, n0 + 16, k0, lane);
  c00 = WMMA_F16(a0, b0, c00);
  c01 = WMMA_F16(a0, b1, c01);
  c10 = WMMA_F16(a1, b0, c10);
  c11 = WMMA_F16(a1, b1, c11);
}

// ---------------------------------------------------------------------------
// f32 -> f16 conversion
// ---------------------------------------------------------------------------
__global__ void k_cvt_f16(const float* __restrict__ src, _Float16* __restrict__ dst, int n) {
  int i = blockIdx.x * blockDim.x + threadIdx.x;
  if (i < n) dst[i] = (_Float16)src[i];
}

// ---------------------------------------------------------------------------
// GEMM1 epilogue element: sigmoid -> sincos -> abc mix -> y[0..3] store
// ---------------------------------------------------------------------------
__device__ __forceinline__ void act_store(float hv, const float* __restrict__ w,
                                          _Float16* __restrict__ Yh,
                                          size_t row, int col) {
  float sg = 1.0f / (1.0f + __expf(-hv));
  float th = 2.0f * PI_F * sg - PI_F;
  float s, c;
  __sincosf(th, &s, &c);
  float sc = s * c;
  float num = 0.f, den = 0.f, av[3];
#pragma unroll
  for (int n = 0; n < 3; ++n) {
    float aa = w[(3*n+0)*3+0]*s + w[(3*n+0)*3+1]*c + w[(3*n+0)*3+2]*sc;
    float bb = w[(3*n+1)*3+0]*s + w[(3*n+1)*3+1]*c + w[(3*n+1)*3+2]*sc;
    float cc = w[(3*n+2)*3+0]*s + w[(3*n+2)*3+1]*c + w[(3*n+2)*3+2]*sc;
    float sb = 1.0f / (1.0f + __expf(-bb));
    num += sb * cc;
    den += sb;
    av[n] = aa > 0.f ? aa : 0.f;
  }
  float f = num / den;
#pragma unroll
  for (int n = 0; n < 3; ++n)
    Yh[((size_t)n * 8192 + row) * 128 + col] = (_Float16)(av[n] * f);
}

// ---------------------------------------------------------------------------
// GEMM1: H = Xh(8192x512) * W0h^T(128x512), fused sincos/abc epilogue
// -> Yh[3][8192][128] f16.  One 32x32 block per wave (2x2 tiles).
// ---------------------------------------------------------------------------
__global__ void __launch_bounds__(128) k_gemm1_act(
    const _Float16* __restrict__ Xh, const _Float16* __restrict__ W0h,
    const float* __restrict__ abcw, _Float16* __restrict__ Yh) {
  const int lane = threadIdx.x & 31;
  const int wave = blockIdx.x * 4 + (threadIdx.x >> 5);
  const int m0 = (wave >> 2) * 32;   // 4 column blocks (N=128)
  const int n0 = (wave & 3) * 32;
  v8f c00 = {}, c01 = {}, c10 = {}, c11 = {};
  for (int k0 = 0; k0 < 512; k0 += 32) {
    __builtin_prefetch(Xh + (size_t)(m0 + (lane & 15)) * 512 + k0 + 32, 0, 1);
    gemm_step_2x2(Xh, 512, m0, W0h, 512, n0, k0, lane, c00, c01, c10, c11);
  }
  float w[27];
#pragma unroll
  for (int i = 0; i < 27; ++i) w[i] = abcw[i];
  const int col = n0 + (lane & 15);
  const int rb  = m0 + ((lane & 16) ? 8 : 0);
#pragma unroll
  for (int v = 0; v < 8; ++v) {
    act_store(c00[v], w, Yh, rb + v,      col);
    act_store(c01[v], w, Yh, rb + v,      col + 16);
    act_store(c10[v], w, Yh, rb + 16 + v, col);
    act_store(c11[v], w, Yh, rb + 16 + v, col + 16);
  }
}

// ---------------------------------------------------------------------------
// GEMM2: QKV[n] = Yh[n](8192x128) * W1h^T(1024x128), head layout out:
// QKVh[n][bh][t][d] f16, bh = b*16+h, d = 0..63.  32x32 block per wave.
// ---------------------------------------------------------------------------
__device__ __forceinline__ void qkv_store(_Float16* __restrict__ QKVh,
                                          int n, int row, int col, float val) {
  const int h = col >> 6, dd = col & 63;
  const int b = row >> 10, t = row & 1023;
  QKVh[((((size_t)n * 128 + b * 16 + h) * 1024 + t) * 64) + dd] = (_Float16)val;
}

__global__ void __launch_bounds__(128) k_gemm2_qkv(
    const _Float16* __restrict__ Yh, const _Float16* __restrict__ W1h,
    _Float16* __restrict__ QKVh) {
  const int n = blockIdx.y;
  const int lane = threadIdx.x & 31;
  const int wave = blockIdx.x * 4 + (threadIdx.x >> 5);
  const int m0 = (wave >> 5) * 32;   // 32 column blocks (N=1024)
  const int n0 = (wave & 31) * 32;
  const _Float16* A = Yh + (size_t)n * 8192 * 128;
  v8f c00 = {}, c01 = {}, c10 = {}, c11 = {};
  for (int k0 = 0; k0 < 128; k0 += 32)
    gemm_step_2x2(A, 128, m0, W1h, 128, n0, k0, lane, c00, c01, c10, c11);
  const int col = n0 + (lane & 15);
  const int rb = m0 + ((lane & 16) ? 8 : 0);
#pragma unroll
  for (int v = 0; v < 8; ++v) {
    qkv_store(QKVh, n, rb + v,      col,      c00[v]);
    qkv_store(QKVh, n, rb + v,      col + 16, c01[v]);
    qkv_store(QKVh, n, rb + 16 + v, col,      c10[v]);
    qkv_store(QKVh, n, rb + 16 + v, col + 16, c11[v]);
  }
}

// ---------------------------------------------------------------------------
// In-place RMS-norm + rotary on q,k planes. NOTE reference quirk:
// theta = head_index * ang (cos_tab indexed by shape[-3] == N_HEAD).
// ---------------------------------------------------------------------------
__global__ void k_norm_rope(_Float16* __restrict__ QKVh) {
  int idx = blockIdx.x * blockDim.x + threadIdx.x;
  if (idx >= 2 * 128 * 1024) return;
  const int n = idx >> 17;
  const int r = idx & ((1 << 17) - 1);
  const int h = (r >> 10) & 15;
  _Float16* row = QKVh + ((size_t)n * 131072 + r) * 64;
  float x[64];
  float ss = 0.f;
#pragma unroll
  for (int i = 0; i < 64; ++i) { x[i] = (float)row[i]; ss += x[i] * x[i]; }
  const float inv = rsqrtf(ss * (1.0f / 64.0f) + RMS_EPS_F);
#pragma unroll
  for (int i = 0; i < 64; ++i) x[i] *= inv;
#pragma unroll
  for (int j = 0; j < 32; ++j) {
    float ang = (j < 16) ? __powf(1.0f / 1024.0f, (float)j * (1.0f / 15.0f)) : 0.0f;
    float th = (float)h * ang;
    float cj = __cosf(th), sj = __sinf(th);
    float x1 = x[j], x2 = x[j + 32];
    row[j]      = (_Float16)( x1 * cj + x2 * sj);
    row[j + 32] = (_Float16)(-x1 * sj + x2 * cj);
  }
}

// V transpose: VTh[bh][d][t] = QKV[2][bh][t][d]  (B operand of attn*V needs V^T)
__global__ void k_vtrans(const _Float16* __restrict__ QKVh, _Float16* __restrict__ VTh) {
  int idx = blockIdx.x * blockDim.x + threadIdx.x;
  if (idx >= 128 * 1024) return;
  const int bh = idx >> 10, t = idx & 1023;
  const _Float16* src = QKVh + ((size_t)2 * 131072 + idx) * 64;
  _Float16* dst = VTh + (size_t)bh * 64 * 1024 + t;
#pragma unroll
  for (int d = 0; d < 64; ++d) dst[(size_t)d * 1024] = src[d];
}

// ---------------------------------------------------------------------------
// Causal flash attention: one wave per 16-row Q tile, 32-key steps.
// 4 WMMA for S (16x32 scores), 4 WMMA for O += P*V. P is relaid C->A layout
// through LDS (same-wave LDS ops are in-order on CDNA5, no barrier needed).
// ---------------------------------------------------------------------------
__global__ void __launch_bounds__(128) k_attention(
    const _Float16* __restrict__ QKVh, const _Float16* __restrict__ VTh,
    _Float16* __restrict__ AOh) {
  __shared__ __align__(16) _Float16 pbuf[4][16 * 32];
  const int lane = threadIdx.x & 31;
  const int wslot = threadIdx.x >> 5;
  const int wave = blockIdx.x * 4 + wslot;
  const int qi = wave & 63;
  const int bh = wave >> 6;
  const int q0 = qi * 16;
  const _Float16* Q  = QKVh + (size_t)bh * 1024 * 64;
  const _Float16* Kk = QKVh + ((size_t)131072 + (size_t)bh * 1024) * 64;
  const _Float16* Vt = VTh + (size_t)bh * 64 * 1024;
  _Float16* pl = pbuf[wslot];

  const v16h qa0 = load_a_frag(Q, 64, q0, 0, lane);
  const v16h qa1 = load_a_frag(Q, 64, q0, 32, lane);

  float Mrow[8], Lrow[8];
  v8f O0 = {}, O1 = {}, O2 = {}, O3 = {};
#pragma unroll
  for (int v = 0; v < 8; ++v) { Mrow[v] = -1.0e30f; Lrow[v] = 0.f; }

  const int colc = lane & 15;
  const int rb = q0 + ((lane & 16) ? 8 : 0);
  const int rl = (lane & 16) ? 8 : 0;
  const int kend = q0 + 16;

  for (int kb = 0; kb < kend; kb += 32) {
    v8f s0 = {}, s1 = {};
    {
      v16h b0 = load_b_frag(Kk, 64, kb, 0, lane);
      s0 = WMMA_F16(qa0, b0, s0);
      v16h b1 = load_b_frag(Kk, 64, kb, 32, lane);
      s0 = WMMA_F16(qa1, b1, s0);
      v16h b2 = load_b_frag(Kk, 64, kb + 16, 0, lane);
      s1 = WMMA_F16(qa0, b2, s1);
      v16h b3 = load_b_frag(Kk, 64, kb + 16, 32, lane);
      s1 = WMMA_F16(qa1, b3, s1);
    }
    float p0[8], p1[8];
#pragma unroll
    for (int v = 0; v < 8; ++v) {
      const int rowg = rb + v;
      float x0 = (kb + colc      <= rowg) ? s0[v] * SCALE_F : -1.0e30f;
      float x1 = (kb + 16 + colc <= rowg) ? s1[v] * SCALE_F : -1.0e30f;
      float mv = fmaxf(x0, x1);
#pragma unroll
      for (int off = 1; off < 16; off <<= 1)
        mv = fmaxf(mv, __shfl_xor(mv, off, 32));   // half-wave row reduction
      const float mnew  = fmaxf(Mrow[v], mv);
      const float alpha = __expf(Mrow[v] - mnew);
      const float e0 = __expf(x0 - mnew);
      const float e1 = __expf(x1 - mnew);
      float lv = e0 + e1;
#pragma unroll
      for (int off = 1; off < 16; off <<= 1)
        lv += __shfl_xor(lv, off, 32);
      Lrow[v] = Lrow[v] * alpha + lv;
      Mrow[v] = mnew;
      O0[v] *= alpha; O1[v] *= alpha; O2[v] *= alpha; O3[v] *= alpha;
      p0[v] = e0; p1[v] = e1;
    }
    // C layout -> A layout through LDS
#pragma unroll
    for (int v = 0; v < 8; ++v) {
      pl[(rl + v) * 32 + colc]      = (_Float16)p0[v];
      pl[(rl + v) * 32 + colc + 16] = (_Float16)p1[v];
    }
    const v16h pa = load_a_frag(pl, 32, 0, 0, lane);
    O0 = WMMA_F16(pa, load_b_frag(Vt, 1024,  0, kb, lane), O0);
    O1 = WMMA_F16(pa, load_b_frag(Vt, 1024, 16, kb, lane), O1);
    O2 = WMMA_F16(pa, load_b_frag(Vt, 1024, 32, kb, lane), O2);
    O3 = WMMA_F16(pa, load_b_frag(Vt, 1024, 48, kb, lane), O3);
  }
  // epilogue: normalize, write AOh[b][t][h*64+d] f16
  const int b = bh >> 4, h = bh & 15;
#pragma unroll
  for (int v = 0; v < 8; ++v) {
    const float inv = 1.0f / Lrow[v];
    const size_t t = rb + v;
    _Float16* dst = AOh + ((size_t)b * 1024 + t) * 1024 + h * 64;
    dst[ 0 + colc] = (_Float16)(O0[v] * inv);
    dst[16 + colc] = (_Float16)(O1[v] * inv);
    dst[32 + colc] = (_Float16)(O2[v] * inv);
    dst[48 + colc] = (_Float16)(O3[v] * inv);
  }
}

// ---------------------------------------------------------------------------
// GEMM3: Rh(8192x128 f16) = AOh(8192x1024) * W2h^T(128x1024). 32x32/wave.
// ---------------------------------------------------------------------------
__global__ void __launch_bounds__(128) k_gemm3(
    const _Float16* __restrict__ AOh, const _Float16* __restrict__ W2h,
    _Float16* __restrict__ Rh) {
  const int lane = threadIdx.x & 31;
  const int wave = blockIdx.x * 4 + (threadIdx.x >> 5);
  const int m0 = (wave >> 2) * 32;   // 4 column blocks (N=128)
  const int n0 = (wave & 3) * 32;
  v8f c00 = {}, c01 = {}, c10 = {}, c11 = {};
  for (int k0 = 0; k0 < 1024; k0 += 32) {
    __builtin_prefetch(AOh + (size_t)(m0 + (lane & 15)) * 1024 + k0 + 32, 0, 1);
    gemm_step_2x2(AOh, 1024, m0, W2h, 1024, n0, k0, lane, c00, c01, c10, c11);
  }
  const int col = n0 + (lane & 15);
  const int rb = m0 + ((lane & 16) ? 8 : 0);
#pragma unroll
  for (int v = 0; v < 8; ++v) {
    Rh[(size_t)(rb + v)      * 128 + col]      = (_Float16)c00[v];
    Rh[(size_t)(rb + v)      * 128 + col + 16] = (_Float16)c01[v];
    Rh[(size_t)(rb + 16 + v) * 128 + col]      = (_Float16)c10[v];
    Rh[(size_t)(rb + 16 + v) * 128 + col + 16] = (_Float16)c11[v];
  }
}

// ---------------------------------------------------------------------------
// GEMM4: out(8192x512 f32) = Rh(8192x128) * W3h^T(512x128). 32x32/wave.
// ---------------------------------------------------------------------------
__global__ void __launch_bounds__(128) k_gemm4(
    const _Float16* __restrict__ Rh, const _Float16* __restrict__ W3h,
    float* __restrict__ out) {
  const int lane = threadIdx.x & 31;
  const int wave = blockIdx.x * 4 + (threadIdx.x >> 5);
  const int m0 = (wave >> 4) * 32;   // 16 column blocks (N=512)
  const int n0 = (wave & 15) * 32;
  v8f c00 = {}, c01 = {}, c10 = {}, c11 = {};
  for (int k0 = 0; k0 < 128; k0 += 32)
    gemm_step_2x2(Rh, 128, m0, W3h, 128, n0, k0, lane, c00, c01, c10, c11);
  const int col = n0 + (lane & 15);
  const int rb = m0 + ((lane & 16) ? 8 : 0);
#pragma unroll
  for (int v = 0; v < 8; ++v) {
    out[(size_t)(rb + v)      * 512 + col]      = c00[v];
    out[(size_t)(rb + v)      * 512 + col + 16] = c01[v];
    out[(size_t)(rb + 16 + v) * 512 + col]      = c10[v];
    out[(size_t)(rb + 16 + v) * 512 + col + 16] = c11[v];
  }
}

// ---------------------------------------------------------------------------
// Host side
// ---------------------------------------------------------------------------
extern "C" void kernel_launch(void* const* d_in, const int* in_sizes, int n_in,
                              void* d_out, int out_size, void* d_ws, size_t ws_size,
                              hipStream_t stream) {
  const float* x      = (const float*)d_in[0];  // (8,1024,512)
  const float* abc_w  = (const float*)d_in[1];  // (9,3)
  const float* lr_w   = (const float*)d_in[2];  // (128,512)
  const float* proj_w = (const float*)d_in[3];  // (1024,128)
  const float* w1     = (const float*)d_in[4];  // (128,1024)
  const float* w2     = (const float*)d_in[5];  // (512,128)
  float* out = (float*)d_out;                   // (8,1024,512)

  char* ws = (char*)d_ws;
  size_t off = 0;
  _Float16* Xh  = (_Float16*)(ws + off); off += (size_t)8192 * 512 * 2;   // 8 MiB
  _Float16* W0h = (_Float16*)(ws + off); off += (size_t)128 * 512 * 2;
  _Float16* W1h = (_Float16*)(ws + off); off += (size_t)1024 * 128 * 2;
  _Float16* W2h = (_Float16*)(ws + off); off += (size_t)128 * 1024 * 2;
  _Float16* W3h = (_Float16*)(ws + off); off += (size_t)512 * 128 * 2;
  _Float16* Yh  = (_Float16*)(ws + off); off += (size_t)3 * 8192 * 128 * 2;   // 6 MiB
  _Float16* QKVh= (_Float16*)(ws + off); off += (size_t)3 * 131072 * 64 * 2;  // 48 MiB
  _Float16* VTh = (_Float16*)(ws + off); off += (size_t)131072 * 64 * 2;      // 16 MiB
  _Float16* AOh = (_Float16*)(ws + off); off += (size_t)8192 * 1024 * 2;      // 16 MiB
  _Float16* Rh  = (_Float16*)(ws + off); off += (size_t)8192 * 128 * 2;       // 2 MiB

  // f32 -> f16 conversions
  k_cvt_f16<<<(4194304 + 255) / 256, 256, 0, stream>>>(x, Xh, 4194304);
  k_cvt_f16<<<(65536   + 255) / 256, 256, 0, stream>>>(lr_w,   W0h, 65536);
  k_cvt_f16<<<(131072  + 255) / 256, 256, 0, stream>>>(proj_w, W1h, 131072);
  k_cvt_f16<<<(131072  + 255) / 256, 256, 0, stream>>>(w1,     W2h, 131072);
  k_cvt_f16<<<(65536   + 255) / 256, 256, 0, stream>>>(w2,     W3h, 65536);

  // GEMM1 + sincos/abc epilogue: (8192/32)*(128/32) = 1024 waves -> 256 blocks
  k_gemm1_act<<<256, 128, 0, stream>>>(Xh, W0h, abc_w, Yh);

  // GEMM2 x3: (8192/32)*(1024/32) = 8192 waves per n -> 2048 blocks, grid.y = 3
  k_gemm2_qkv<<<dim3(2048, 3), 128, 0, stream>>>(Yh, W1h, QKVh);

  // RMS-norm + rotary (q,k in place), V transpose
  k_norm_rope<<<(262144 + 255) / 256, 256, 0, stream>>>(QKVh);
  k_vtrans<<<(131072 + 255) / 256, 256, 0, stream>>>(QKVh, VTh);

  // Attention: 128 bh * 64 q-tiles = 8192 waves -> 2048 blocks
  k_attention<<<2048, 128, 0, stream>>>(QKVh, VTh, AOh);

  // GEMM3: (8192/32)*(128/32) = 1024 waves -> 256 blocks
  k_gemm3<<<256, 128, 0, stream>>>(AOh, W2h, Rh);

  // GEMM4: (8192/32)*(512/32) = 4096 waves -> 1024 blocks, writes f32 output
  k_gemm4<<<1024, 128, 0, stream>>>(Rh, W3h, out);

  (void)in_sizes; (void)n_in; (void)out_size; (void)ws_size;
}
